// ClusterPrior_19842748907739
// MI455X (gfx1250) — matrix operand order repaired and checked by reading, not compiled, not run on hardware
//
#include <hip/hip_runtime.h>

typedef __attribute__((ext_vector_type(16))) __bf16 v16bf;
typedef __attribute__((ext_vector_type(8)))  float  v8f;
typedef __attribute__((ext_vector_type(4)))  float  f4;
typedef __attribute__((ext_vector_type(4)))  unsigned int u4;

#define D_DIM 128
#define K_CENT 1024
#define KT (K_CENT / 16)      // 64 centroid tiles of 16
#define TOTAL_ROWS (8 * 4096) // B*N

// ---------------------------------------------------------------------------
// Kernel 0: fold standardization into centroids, split into hi/lo bf16, and
// compute per-centroid bias:  bias_k = 0.5*||c_k||^2 + sum_d mean_d * c'_kd
// where c'_kd = c_kd / scale_d.  argmin_k d2 == argmin_k (bias_k - x . c'_k).
// ---------------------------------------------------------------------------
__global__ __launch_bounds__(128) void ClusterPrior_prep(
    const float* __restrict__ centroids, const float* __restrict__ mean,
    const float* __restrict__ scale,
    unsigned short* __restrict__ cphi, unsigned short* __restrict__ cplo,
    float* __restrict__ bias)
{
  int k = blockIdx.x * blockDim.x + threadIdx.x;
  if (k >= K_CENT) return;
  const float* cr = centroids + (long)k * D_DIM;
  unsigned short* ph = cphi + (long)k * D_DIM;
  unsigned short* pl = cplo + (long)k * D_DIM;
  float sumc2 = 0.f, off = 0.f;
  for (int d = 0; d < D_DIM; ++d) {
    float c  = cr[d];
    float cp = c / scale[d];
    sumc2 += c * c;
    off   += mean[d] * cp;
    __bf16 h = (__bf16)cp;                 // RNE high part
    __bf16 l = (__bf16)(cp - (float)h);    // residual low part
    ph[d] = __builtin_bit_cast(unsigned short, h);
    pl[d] = __builtin_bit_cast(unsigned short, l);
  }
  bias[k] = 0.5f * sumc2 + off;
}

// ---------------------------------------------------------------------------
// Kernel 1: per-wave 16-point tile vs all 1024 centroids via bf16 WMMA
// (hi/lo split for ~fp32-equivalent precision), argmin tracking, one-hot
// streaming store.
// ---------------------------------------------------------------------------
union BV { u4 q[2]; v16bf v; };

__global__ __launch_bounds__(256) void ClusterPrior_assign(
    const float* __restrict__ X, const float* __restrict__ mask,
    const unsigned short* __restrict__ cphi,
    const unsigned short* __restrict__ cplo,
    const float* __restrict__ bias, float* __restrict__ out)
{
  const int lane   = threadIdx.x & 31;
  const int wave   = threadIdx.x >> 5;
  const int laneLo = lane & 15;      // column / row-in-tile selector
  const int laneHi = lane >> 4;      // half-wave K-split selector
  const long row0 = (long)blockIdx.x * 128 + (long)wave * 16;
  const long rowA = row0 + laneLo;

  // --- Load A operand: 16 rows x 128 dims, contiguous k order (same as B):
  // lane<16 holds row=laneLo, dims {base..base+15};
  // lane>=16 holds same row, dims {base+16..base+31}.
  v16bf a_hi[4], a_lo[4];
  const float* xr = X + rowA * D_DIM;
  #pragma unroll
  for (int c = 0; c < 4; ++c) {
    const int base = c * 32 + laneHi * 16;
    float f[16];
    #pragma unroll
    for (int j = 0; j < 8; ++j) f[j]     = xr[base + j];
    #pragma unroll
    for (int j = 0; j < 8; ++j) f[8 + j] = xr[base + 8 + j];
    #pragma unroll
    for (int i = 0; i < 16; ++i) {
      __bf16 h = (__bf16)f[i];
      a_hi[c][i] = h;
      a_lo[c][i] = (__bf16)(f[i] - (float)h);
    }
  }

  float minv[8];
  int   mini[8];
  #pragma unroll
  for (int r = 0; r < 8; ++r) { minv[r] = __builtin_inff(); mini[r] = 0; }

  // --- Sweep 64 centroid tiles of 16 columns each.
  for (int t = 0; t < KT; ++t) {
    const int kcol = t * 16 + laneLo;                       // this lane's column
    const unsigned short* bh_base = cphi + (long)kcol * D_DIM;
    const unsigned short* bl_base = cplo + (long)kcol * D_DIM;
    if (t + 1 < KT) {
      __builtin_prefetch(cphi + (long)(kcol + 16) * D_DIM, 0, 1);
      __builtin_prefetch(cplo + (long)(kcol + 16) * D_DIM, 0, 1);
    }

    v8f acc = {};
    #pragma unroll
    for (int c = 0; c < 4; ++c) {
      // B layout (32x16 bf16): lane<16 -> K=0..15, lane>=16 -> K=16..31
      // of column N = lane%16, consecutive dims packed 2-per-VGPR.
      const int boff = c * 32 + laneHi * 16;
      BV bh, bl;
      bh.q[0] = *(const u4*)(bh_base + boff);
      bh.q[1] = *(const u4*)(bh_base + boff + 8);
      bl.q[0] = *(const u4*)(bl_base + boff);
      bl.q[1] = *(const u4*)(bl_base + boff + 8);
      acc = __builtin_amdgcn_wmma_f32_16x16x32_bf16(false, a_hi[c], false, bh.v,
                                                    (short)0, acc, false, false);
      acc = __builtin_amdgcn_wmma_f32_16x16x32_bf16(false, a_lo[c], false, bh.v,
                                                    (short)0, acc, false, false);
      acc = __builtin_amdgcn_wmma_f32_16x16x32_bf16(false, a_hi[c], false, bl.v,
                                                    (short)0, acc, false, false);
    }

    // score = bias_k - cross;  C layout: lane holds column kcol for 8 rows.
    const float bk = bias[kcol];
    #pragma unroll
    for (int r = 0; r < 8; ++r) {
      const float s = bk - acc[r];
      if (s < minv[r]) { minv[r] = s; mini[r] = kcol; }
    }
  }

  // --- Reduce (min,idx) across the 16 lanes holding each row.
  // Rows 0..7 live in lanes 0..15 (VGPR r); rows 8..15 in lanes 16..31.
  #pragma unroll
  for (int r = 0; r < 8; ++r) {
    float v = minv[r];
    int   i = mini[r];
    #pragma unroll
    for (int m = 1; m < 16; m <<= 1) {
      const float ov = __shfl_xor(v, m, 16);
      const int   oi = __shfl_xor(i, m, 16);
      if (ov < v || (ov == v && oi < i)) { v = ov; i = oi; }  // first-min wins
    }
    mini[r] = i;
  }

  // --- Stream one-hot rows: 16 rows x 1024 floats, value computed inline.
  #pragma unroll
  for (int m = 0; m < 16; ++m) {
    const int am = __shfl(mini[m & 7], (m >> 3) << 4, 32);   // argmin for row m
    const float mv = mask[row0 + m];
    float* orow = out + (row0 + m) * (long)K_CENT;
    #pragma unroll
    for (int kk = 0; kk < 8; ++kk) {
      const int k0 = kk * 128 + lane * 4;
      f4 val;
      val.x = (k0 + 0 == am) ? mv : 0.f;
      val.y = (k0 + 1 == am) ? mv : 0.f;
      val.z = (k0 + 2 == am) ? mv : 0.f;
      val.w = (k0 + 3 == am) ? mv : 0.f;
      __builtin_nontemporal_store(val, (f4*)(orow + k0));
    }
  }
}

extern "C" void kernel_launch(void* const* d_in, const int* in_sizes, int n_in,
                              void* d_out, int out_size, void* d_ws, size_t ws_size,
                              hipStream_t stream) {
  const float* X         = (const float*)d_in[0];
  const float* mask      = (const float*)d_in[1];
  const float* centroids = (const float*)d_in[2];
  const float* mean      = (const float*)d_in[3];
  const float* scale     = (const float*)d_in[4];
  float* out = (float*)d_out;

  unsigned short* cphi = (unsigned short*)d_ws;                 // 256 KB
  unsigned short* cplo = cphi + (size_t)K_CENT * D_DIM;         // 256 KB
  float* bias = (float*)(cplo + (size_t)K_CENT * D_DIM);        //   4 KB

  hipLaunchKernelGGL(ClusterPrior_prep, dim3(K_CENT / 128), dim3(128), 0, stream,
                     centroids, mean, scale, cphi, cplo, bias);
  hipLaunchKernelGGL(ClusterPrior_assign, dim3(TOTAL_ROWS / 128), dim3(256), 0, stream,
                     X, mask, cphi, cplo, bias, out);
}
